// PurePyTorchINT4Linear_44006234915287
// MI455X (gfx1250) — matrix operand order repaired and checked by reading, not compiled, not run on hardware
//
#include <hip/hip_runtime.h>

typedef __attribute__((ext_vector_type(16))) _Float16 v16h;
typedef __attribute__((ext_vector_type(8)))  _Float16 v8h;
typedef __attribute__((ext_vector_type(4)))  _Float16 h4;
typedef __attribute__((ext_vector_type(2)))  _Float16 h2;
typedef __attribute__((ext_vector_type(8)))  float    v8f;
typedef __attribute__((ext_vector_type(4)))  unsigned u32x4;
typedef __attribute__((ext_vector_type(4)))  int      i32x4;
typedef __attribute__((ext_vector_type(8)))  int      i32x8;

#define BLOCK_M 256
#define BLOCK_N 128
#define BIG_K   128            // staging depth == quant group size
#define LSTRIDE 136            // 128 + 8 pad halves; 272B rows: 16B-aligned, bank-conflict-free

// ---- CDNA5 Tensor Data Mover plumbing (guarded; falls back to sync staging) ----
#if defined(__AMDGCN__)
#if __has_builtin(__builtin_amdgcn_tensor_load_to_lds)
#define USE_TDM 1
// amdgpu-toolchain (clang-23) 6-arg form:
//   (uint32x4 g0, int32x8 g1, int32x4 g2, int32x4 g3, int32x8 pad, i32 cpol)
#define TDM_ISSUE(g0, g1)                                                  \
    do {                                                                   \
        i32x4 _z4 = {0, 0, 0, 0};                                          \
        i32x8 _z8 = {0, 0, 0, 0, 0, 0, 0, 0};                              \
        __builtin_amdgcn_tensor_load_to_lds(g0, g1, _z4, _z4, _z8, 0);     \
    } while (0)
#endif
#if __has_builtin(__builtin_amdgcn_s_wait_tensorcnt)
#define TDM_WAIT() __builtin_amdgcn_s_wait_tensorcnt(0)
#else
#define TDM_WAIT() asm volatile("s_wait_tensorcnt 0x0" ::: "memory")
#endif
#endif

#ifdef USE_TDM
// 2D tile load: BLOCK_M rows x BIG_K halves, row-padded to LSTRIDE halves in LDS.
__device__ __forceinline__ void tdm_issue_x(const _Float16* gsrc, unsigned lds_byte_off,
                                            int M, int K) {
    unsigned long long ga = (unsigned long long)(uintptr_t)gsrc;  // tile start (bytes)
    u32x4 g0;
    g0[0] = 1u;                                          // count=1, user descriptor
    g0[1] = lds_byte_off;                                // LDS destination
    g0[2] = (unsigned)ga;                                // global_addr[31:0]
    g0[3] = (unsigned)((ga >> 32) & 0x1FFFFFFu)          // global_addr[56:32]
          | (2u << 30);                                  // type = 2 (image)
    i32x8 g1;
    g1[0] = (int)((1u << 16)      // data_size = 2 bytes
                | (1u << 20)      // pad_enable
                | (5u << 22)      // pad_interval: 64 DWORDs (= 128 halves of data)
                | (3u << 25));    // pad_amount:   4 DWORDs  (-> LSTRIDE = 136 halves)
    g1[1] = (int)(((unsigned)K & 0xFFFFu) << 16);                    // tensor_dim0 lo16
    g1[2] = (int)((((unsigned)K >> 16) & 0xFFFFu)
                | (((unsigned)M & 0xFFFFu) << 16));                  // dim0 hi | dim1 lo
    g1[3] = (int)((((unsigned)M >> 16) & 0xFFFFu)
                | ((unsigned)BIG_K << 16));                          // dim1 hi | tile_dim0
    g1[4] = BLOCK_M;                                                 // tile_dim1 | tile_dim2=0
    g1[5] = K;                                                       // tensor_dim0_stride lo32
    g1[6] = 0;                                                       // stride hi | dim1_stride lo
    g1[7] = 0;
    TDM_ISSUE(g0, g1);
}
#endif

// W4A16 GEMM: out[M,N] = x[M,K] * dequant(qweight,scales,qzeros)[K,N]
__global__ __launch_bounds__(256)
void int4_wmma_gemm(const _Float16* __restrict__ x,
                    const int*      __restrict__ qw,
                    const _Float16* __restrict__ scales,
                    const int*      __restrict__ qzeros,
                    float*          __restrict__ out,
                    int M, int N, int K)
{
    __shared__ _Float16 lds_x[2][BLOCK_M * LSTRIDE];   // [row][k], double-buffered
    __shared__ _Float16 lds_w[2][BLOCK_N * LSTRIDE];   // [col][k], double-buffered

    const int tid  = threadIdx.x;
    const int lane = tid & 31;
    const int wave = tid >> 5;
    const int wm   = wave >> 1;          // 0..3 : 64 rows each
    const int wn   = wave & 1;           // 0..1 : 64 cols each
    const int half = lane >> 4;
    const int l15  = lane & 15;

    const int row0 = blockIdx.y * BLOCK_M;
    const int col0 = blockIdx.x * BLOCK_N;

    // dequant map: thread -> 4 consecutive columns, 8 packed rows (of 64 per stage)
    const int dn0 = (tid & 31) * 4;
    const int kr0 = (tid >> 5) * 8;
    const int nc0 = col0 + dn0;

    v8f acc[4][4] = {};                  // 64x64 wave tile: 4 m-tiles x 4 n-tiles
    const int nbig = K / BIG_K;

#ifdef USE_TDM
    if (wave == 0)
        tdm_issue_x(x + (size_t)row0 * K, (unsigned)(size_t)&lds_x[0][0], M, K);
#endif

    for (int bi = 0; bi < nbig; ++bi) {
        const int buf = bi & 1;
        _Float16*       lw = lds_w[buf];
        const _Float16* lx = lds_x[buf];

        // ---- per-group dequant constants (group index == stage index) ----
        h4    sc4 = *(const h4*)(scales + (size_t)bi * N + nc0);
        i32x4 zq4 = *(const i32x4*)(qzeros + (size_t)(bi >> 1) * N + nc0);
        const int zsh = (bi & 1) * 4;
        h2 s2[4], b2[4];
#pragma unroll
        for (int c = 0; c < 4; ++c) {
            _Float16 s  = sc4[c];
            int      zn = (zq4[c] >> zsh) & 15;
            _Float16 zb = (_Float16)(-(float)s * (float)(1024 + zn));
            s2[c][0] = s;  s2[c][1] = s;
            b2[c][0] = zb; b2[c][1] = zb;
        }

        // ---- cooperative dequant: 8 x b128 qweight loads -> pk_fma -> LDS ----
        const int* qbase = qw + (size_t)(bi * (BIG_K / 2) + kr0) * N + nc0;
#pragma unroll
        for (int j = 0; j < 8; ++j) {
            i32x4 q = *(const i32x4*)(qbase + (size_t)j * N);
#pragma unroll
            for (int c = 0; c < 4; ++c) {
                // {1024+w_even, 1024+w_odd} packed f16; w = s*(wq - z) via pk_fma
                unsigned p = ((unsigned)q[c] & 0xFu)
                           | (((unsigned)q[c] & 0xF0u) << 12) | 0x64006400u;
                h2 hv = __builtin_bit_cast(h2, p);
                h2 w  = hv * s2[c] + b2[c];
                *(h2*)(lw + (dn0 + c) * LSTRIDE + 2 * (kr0 + j)) = w;
            }
        }
        if (bi + 1 < nbig)
            __builtin_prefetch(qbase + (size_t)(BIG_K / 2) * N, 0, 1);

#ifdef USE_TDM
        if (wave == 0) TDM_WAIT();         // x tile for this stage landed in LDS
#else
        {   // fallback: sync staging of x tile (one row = 16 x b128 per thread)
            const _Float16* src = x + (size_t)(row0 + tid) * K + bi * BIG_K;
            _Float16*       dst = lds_x[buf] + tid * LSTRIDE;
#pragma unroll
            for (int c8 = 0; c8 < 16; ++c8)
                *(v8h*)(dst + c8 * 8) = *(const v8h*)(src + c8 * 8);
        }
#endif

        __syncthreads();   // single barrier per 128-K stage (double buffering)

#ifdef USE_TDM
        if (wave == 0 && bi + 1 < nbig)    // overlap next stage's DMA with WMMA
            tdm_issue_x(x + (size_t)row0 * K + (bi + 1) * BIG_K,
                        (unsigned)(size_t)&lds_x[buf ^ 1][0], M, K);
#endif

        // ---- 4 k-steps x (4 m-tiles x 4 n-tiles) WMMA ----
#pragma unroll
        for (int kk = 0; kk < 4; ++kk) {
            const int ko = kk * 32;
            v16h bfrag[4];
#pragma unroll
            for (int nt = 0; nt < 4; ++nt) {
                const _Float16* bp =
                    lw + (wn * 64 + nt * 16 + l15) * LSTRIDE + ko + half * 8;
                v8h blo = *(const v8h*)bp;          // K = kb .. kb+7
                v8h bhi = *(const v8h*)(bp + 16);   // K = kb+16 .. kb+23
                bfrag[nt] = __builtin_shufflevector(blo, bhi,
                    0,1,2,3,4,5,6,7,8,9,10,11,12,13,14,15);
            }
#pragma unroll
            for (int mt = 0; mt < 4; ++mt) {
                const _Float16* ap =
                    lx + (wm * 64 + mt * 16 + l15) * LSTRIDE + ko + half * 8;
                v8h alo = *(const v8h*)ap;          // K = kb .. kb+7
                v8h ahi = *(const v8h*)(ap + 16);   // K = kb+16 .. kb+23
                v16h afrag = __builtin_shufflevector(alo, ahi,
                    0,1,2,3,4,5,6,7,8,9,10,11,12,13,14,15);
#pragma unroll
                for (int nt = 0; nt < 4; ++nt)
                    acc[mt][nt] = __builtin_amdgcn_wmma_f32_16x16x32_f16(
                        false, afrag, false, bfrag[nt],
                        (short)0, acc[mt][nt], false, false);
            }
        }
    }

    // ---- write back C (f32): VGPR r -> M = r + half*8, lane l15 -> N ----
#pragma unroll
    for (int mt = 0; mt < 4; ++mt) {
#pragma unroll
        for (int nt = 0; nt < 4; ++nt) {
            const int rbase = row0 + wm * 64 + mt * 16 + half * 8;
            const int c     = col0 + wn * 64 + nt * 16 + l15;
#pragma unroll
            for (int r = 0; r < 8; ++r)
                out[(size_t)(rbase + r) * N + c] = acc[mt][nt][r];
        }
    }
}

extern "C" void kernel_launch(void* const* d_in, const int* in_sizes, int n_in,
                              void* d_out, int out_size, void* d_ws, size_t ws_size,
                              hipStream_t stream) {
    const _Float16* x  = (const _Float16*)d_in[0];
    const int*      qw = (const int*)d_in[1];
    const _Float16* sc = (const _Float16*)d_in[2];
    const int*      qz = (const int*)d_in[3];
    float*          out = (float*)d_out;

    const int K = 4096;
    const int N = 11008;
    const int M = in_sizes[0] / K;   // 8192

    dim3 grid(N / BLOCK_N, M / BLOCK_M);   // 86 x 32
    dim3 block(256);
    int4_wmma_gemm<<<grid, block, 0, stream>>>(x, qw, sc, qz, out, M, N, K);
}
